// Transformer_16441134809839
// MI455X (gfx1250) — matrix operand order repaired and checked by reading, not compiled, not run on hardware
//
#include <hip/hip_runtime.h>

// ---- problem constants (from reference) ----
#define DIMX   512
#define HEADS  8
#define DHEAD  64
#define SEQ    2305
#define BATCH  2
#define ROWS   (BATCH * SEQ)     // 4610
#define GHW    48                // GH == GW
#define ALEN   49                // 1 BOS + 48 axial tokens
#define QSCALE 0.125f            // DHEAD^-0.5
#define WELEMS (DIMX * DIMX)     // elements per weight matrix

typedef __attribute__((ext_vector_type(16))) _Float16 v16h;
typedef __attribute__((ext_vector_type(8)))  _Float16 v8h;
typedef __attribute__((ext_vector_type(8)))  float    v8f;

// ---- WMMA fragment builders (CDNA5 ISA 7.12.2, wave32, 16-bit, 16x16x32) ----
// A (16x32): lane holds row M = lane%16; elems 0..7 <-> K(8*hs .. 8*hs+7),
//            elems 8..15 <-> K(16+8*hs .. 23+8*hs).  => two contiguous b128 loads.
// B (32x16): lane holds col N = lane%16; elems 0..7 <-> K(8*hs .. 8*hs+7),
//            elems 8..15 <-> K(16+8*hs .. 23+8*hs).  => two contiguous b128 loads.
__device__ __forceinline__ v16h frag_cat(v8h lo, v8h hi) {
    return __builtin_shufflevector(lo, hi, 0, 1, 2, 3, 4, 5, 6, 7,
                                           8, 9, 10, 11, 12, 13, 14, 15);
}
__device__ __forceinline__ v16h load_a_frag(const _Float16* p, int hs) {
    v8h lo = *(const v8h*)(p + 8 * hs);
    v8h hi = *(const v8h*)(p + 16 + 8 * hs);
    return frag_cat(lo, hi);
}
__device__ __forceinline__ v16h load_b_frag(const _Float16* p, int hs) {
    v8h lo = *(const v8h*)(p + 8 * hs);
    v8h hi = *(const v8h*)(p + 16 + 8 * hs);
    return frag_cat(lo, hi);
}

__device__ __forceinline__ v8f wmma_f16(v16h a, v16h b, v8f c) {
    return __builtin_amdgcn_wmma_f32_16x16x32_f16(false, a, false, b, (short)0, c,
                                                  false, false);
}

// butterfly reductions within 16-lane halves (masks < 16 never cross halves)
__device__ __forceinline__ float half_max(float v) {
    v = fmaxf(v, __shfl_xor(v, 1, 32));
    v = fmaxf(v, __shfl_xor(v, 2, 32));
    v = fmaxf(v, __shfl_xor(v, 4, 32));
    v = fmaxf(v, __shfl_xor(v, 8, 32));
    return v;
}
__device__ __forceinline__ float half_sum(float v) {
    v += __shfl_xor(v, 1, 32);
    v += __shfl_xor(v, 2, 32);
    v += __shfl_xor(v, 4, 32);
    v += __shfl_xor(v, 8, 32);
    return v;
}

// ============================================================================
// One-time conversions: f32 -> f16, weights also transposed to [N][K] so that
// B fragments are contiguous 128-bit loads.
// ============================================================================
__global__ __launch_bounds__(256)
void cvt_w_t(const float* __restrict__ W, _Float16* __restrict__ Wt) {
    int idx = blockIdx.x * 256 + threadIdx.x;       // idx = n*512 + k
    if (idx < WELEMS) {
        int n = idx >> 9, k = idx & 511;
        Wt[idx] = (_Float16)W[(size_t)k * DIMX + n];
    }
}
__global__ __launch_bounds__(256)
void cvt_x(const float* __restrict__ X, _Float16* __restrict__ xh) {
    size_t idx = (size_t)blockIdx.x * 256 + threadIdx.x;
    if (idx < (size_t)ROWS * DIMX) xh[idx] = (_Float16)X[idx];
}

// ============================================================================
// Fused QKV projection: q = (x@Wq)*scale, k = x@Wk, v = x@Wv  (f16 in/out)
// One wave computes a 32x16 output tile; 2 A-frags shared by 3 weights
// => 6 WMMAs per 10 b128 loads per K-step. V is stored TRANSPOSED:
// vt[((b*HEADS+h)*DHEAD+d)*SEQ + tok] so attention PV B-frags are contiguous.
// ============================================================================
__global__ __launch_bounds__(128)
void qkv_gemm(const _Float16* __restrict__ xh,
              const _Float16* __restrict__ Wtq, const _Float16* __restrict__ Wtk,
              const _Float16* __restrict__ Wtv,
              _Float16* __restrict__ qf, _Float16* __restrict__ kf,
              _Float16* __restrict__ vt) {
    const int lane = threadIdx.x & 31, wid = threadIdx.x >> 5;
    const int hs = lane >> 4, ln = lane & 15;
    const int ct = blockIdx.x * 4 + wid;             // 0..31 col tiles (16 wide)
    const int r0 = blockIdx.y * 32;                  // row tile (32 tall)
    const int col = ct * 16 + ln;

    const _Float16* a0p = xh + (size_t)min(r0 + ln,      ROWS - 1) * DIMX;
    const _Float16* a1p = xh + (size_t)min(r0 + 16 + ln, ROWS - 1) * DIMX;
    const _Float16* bqp = Wtq + (size_t)col * DIMX;
    const _Float16* bkp = Wtk + (size_t)col * DIMX;
    const _Float16* bvp = Wtv + (size_t)col * DIMX;

    v8f acc[6] = {};                                 // [q0,q1,k0,k1,v0,v1]
    for (int kk = 0; kk < DIMX; kk += 32) {
        __builtin_prefetch(a0p + kk + 128, 0, 3);    // global_prefetch_b8
        __builtin_prefetch(bqp + kk + 128, 0, 3);
        v16h a0 = load_a_frag(a0p + kk, hs);
        v16h a1 = load_a_frag(a1p + kk, hs);
        v16h bq = load_b_frag(bqp + kk, hs);
        v16h bk = load_b_frag(bkp + kk, hs);
        v16h bv = load_b_frag(bvp + kk, hs);
        acc[0] = wmma_f16(a0, bq, acc[0]);
        acc[1] = wmma_f16(a1, bq, acc[1]);
        acc[2] = wmma_f16(a0, bk, acc[2]);
        acc[3] = wmma_f16(a1, bk, acc[3]);
        acc[4] = wmma_f16(a0, bv, acc[4]);
        acc[5] = wmma_f16(a1, bv, acc[5]);
    }
    const int h = col >> 6, d = col & 63;            // head / feature for vt
#pragma unroll
    for (int t2 = 0; t2 < 2; ++t2)
#pragma unroll
        for (int r = 0; r < 8; ++r) {
            int row = r0 + t2 * 16 + r + hs * 8;     // C layout: M = r + 8*hs
            if (row < ROWS) {
                size_t o = (size_t)row * DIMX + col;
                qf[o] = (_Float16)(acc[t2][r] * QSCALE);
                kf[o] = (_Float16)acc[2 + t2][r];
                int b = row / SEQ, tok = row % SEQ;
                vt[((size_t)(b * HEADS + h) * DHEAD + d) * SEQ + tok] =
                    (_Float16)acc[4 + t2][r];
            }
        }
}

// ============================================================================
// Output projection + bias + residual:  X += A @ Wo + bo  (A is f16 aoh)
// One wave computes 32x32 => 4 WMMAs per 8 b128 loads per K-step.
// ============================================================================
__global__ __launch_bounds__(128)
void out_proj(const _Float16* __restrict__ A, const _Float16* __restrict__ Wt,
              const float* __restrict__ bo, float* __restrict__ X) {
    const int lane = threadIdx.x & 31, wid = threadIdx.x >> 5;
    const int hs = lane >> 4, ln = lane & 15;
    const int c0 = (blockIdx.x * 4 + wid) * 32;      // 0..480 col tiles (32 wide)
    const int r0 = blockIdx.y * 32;                  // row tile (32 tall)

    const _Float16* a0p = A + (size_t)min(r0 + ln,      ROWS - 1) * DIMX;
    const _Float16* a1p = A + (size_t)min(r0 + 16 + ln, ROWS - 1) * DIMX;
    const _Float16* b0p = Wt + (size_t)(c0 + ln) * DIMX;
    const _Float16* b1p = Wt + (size_t)(c0 + 16 + ln) * DIMX;

    v8f acc[4] = {};                                 // [r0c0, r0c1, r1c0, r1c1]
    for (int kk = 0; kk < DIMX; kk += 32) {
        __builtin_prefetch(a0p + kk + 128, 0, 3);
        v16h a0 = load_a_frag(a0p + kk, hs);
        v16h a1 = load_a_frag(a1p + kk, hs);
        v16h b0 = load_b_frag(b0p + kk, hs);
        v16h b1 = load_b_frag(b1p + kk, hs);
        acc[0] = wmma_f16(a0, b0, acc[0]);
        acc[1] = wmma_f16(a0, b1, acc[1]);
        acc[2] = wmma_f16(a1, b0, acc[2]);
        acc[3] = wmma_f16(a1, b1, acc[3]);
    }
#pragma unroll
    for (int ri = 0; ri < 2; ++ri)
#pragma unroll
        for (int ci = 0; ci < 2; ++ci) {
            int col = c0 + ci * 16 + ln;
            float bias = bo[col];
#pragma unroll
            for (int r = 0; r < 8; ++r) {
                int row = r0 + ri * 16 + r + hs * 8;
                if (row < ROWS)
                    X[(size_t)row * DIMX + col] += acc[ri * 2 + ci][r] + bias;
            }
        }
}

// ============================================================================
// Axial attention (typ 0: rows, typ 1: cols). One block = (b, h, group n).
// Sequence = [BOS] + 48 tokens, causal. Tiny FLOPs -> LDS-staged VALU softmax.
// ============================================================================
__device__ __forceinline__ int axial_tok(int typ, int n, int j) {
    return (j == 0) ? 0 : (typ == 0 ? 1 + n * GHW + (j - 1) : 1 + (j - 1) * GHW + n);
}

__global__ __launch_bounds__(64)
void axial_attn(const _Float16* __restrict__ qf, const _Float16* __restrict__ kf,
                const _Float16* __restrict__ vt, _Float16* __restrict__ aoh,
                int typ) {
    __shared__ _Float16 Qh[ALEN * DHEAD];
    __shared__ _Float16 Kh[ALEN * DHEAD];
    __shared__ _Float16 Vh[ALEN * DHEAD];
    __shared__ float    S[ALEN * (ALEN + 1)];

    const int t = threadIdx.x;                 // 0..63 = feature lane
    const int n = blockIdx.x, h = blockIdx.y, b = blockIdx.z;
    const size_t bb = (size_t)b * SEQ;
    const _Float16* vrow = vt + ((size_t)(b * HEADS + h) * DHEAD + t) * SEQ;

    for (int j = 0; j < ALEN; ++j) {
        int tok = axial_tok(typ, n, j);
        size_t g = (bb + tok) * DIMX + h * DHEAD + t;
        Qh[j * DHEAD + t] = qf[g];
        Kh[j * DHEAD + t] = kf[g];
        Vh[j * DHEAD + t] = vrow[tok];
    }
    __syncthreads();

    if (t < ALEN) {                            // thread t owns query row i=t
        const int i = t;
        float m = -1e30f;
        for (int j = 0; j <= i; ++j) {         // causal
            float s = 0.f;
            for (int d = 0; d < DHEAD; ++d)
                s += (float)Qh[i * DHEAD + d] * (float)Kh[j * DHEAD + d];
            S[i * (ALEN + 1) + j] = s;
            m = fmaxf(m, s);
        }
        float l = 0.f;
        for (int j = 0; j <= i; ++j) {
            float p = __expf(S[i * (ALEN + 1) + j] - m);
            S[i * (ALEN + 1) + j] = p;
            l += p;
        }
        float inv = 1.f / l;
        for (int j = 0; j <= i; ++j) S[i * (ALEN + 1) + j] *= inv;
    }
    __syncthreads();

    for (int i = 0; i < ALEN; ++i) {           // thread t = output feature
        if (i == 0 && n != 0) continue;        // BOS output comes from group 0 only
        float acc = 0.f;
        for (int j = 0; j <= i; ++j)
            acc += S[i * (ALEN + 1) + j] * (float)Vh[j * DHEAD + t];
        aoh[(bb + axial_tok(typ, n, i)) * DIMX + h * DHEAD + t] = (_Float16)acc;
    }
}

// ============================================================================
// Nearby (masked full) attention: flash-attention, one wave per 16-query tile
// per (b,h). 32 keys per iteration: S via WMMA, online softmax in C-layout,
// P through LDS to re-fragment into A-layout, O += P@V via WMMA (V transposed
// so its B-fragments are contiguous b128 loads).
// ============================================================================
__global__ __launch_bounds__(32)
void nearby_attn(const _Float16* __restrict__ qf, const _Float16* __restrict__ kf,
                 const _Float16* __restrict__ vt,
                 const unsigned char* __restrict__ mask,
                 _Float16* __restrict__ aoh) {
    __shared__ _Float16 Plds[16 * 32];

    const int lane = threadIdx.x;
    const int hs = lane >> 4, ln = lane & 15;
    const int qt = blockIdx.x, h = blockIdx.y, b = blockIdx.z;
    const size_t bb = (size_t)b * SEQ;

    // Q fragments (A layout, rows qt*16+ln), feature halves d[0..31], d[32..63]
    const int qrow = min(qt * 16 + ln, SEQ - 1);
    const _Float16* qp = qf + (bb + qrow) * DIMX + h * DHEAD;
    v16h qa0 = load_a_frag(qp, hs);
    v16h qa1 = load_a_frag(qp + 32, hs);

    // V rows (transposed): one pointer per feature col-tile
    const _Float16* vr[4];
#pragma unroll
    for (int ct = 0; ct < 4; ++ct)
        vr[ct] = vt + ((size_t)(b * HEADS + h) * DHEAD + ct * 16 + ln) * SEQ;

    v8f oacc[4] = {};                          // O tile 16x64 (4 col-tiles)
    float mrow[8], lrow[8];                    // row stats; slot r <-> row r+8*hs
#pragma unroll
    for (int r = 0; r < 8; ++r) { mrow[r] = -1e30f; lrow[r] = 0.f; }

    for (int kb = 0; kb < SEQ; kb += 32) {
        // --- scores: two 16-key subtiles ---
        v8f s[2];
#pragma unroll
        for (int st = 0; st < 2; ++st) {
            int krow = min(kb + st * 16 + ln, SEQ - 1);
            const _Float16* kp = kf + (bb + krow) * DIMX + h * DHEAD;
            v16h b0 = load_b_frag(kp, hs);       // contraction = feature d
            v16h b1 = load_b_frag(kp + 32, hs);
            v8f c = {};
            c = wmma_f16(qa0, b0, c);
            c = wmma_f16(qa1, b1, c);
            s[st] = c;
        }

        // --- mask + online softmax (row R = qt*16 + r + 8*hs, col = kb+st*16+ln)
#pragma unroll
        for (int r = 0; r < 8; ++r) {
            int row = qt * 16 + r + hs * 8;
            int rm  = min(row, SEQ - 1);
            float s0 = s[0][r], s1 = s[1][r];
            int c0 = kb + ln, c1 = kb + 16 + ln;
            if (c0 >= SEQ || mask[(size_t)rm * SEQ + c0]) s0 = -1e30f;
            if (c1 >= SEQ || mask[(size_t)rm * SEQ + c1]) s1 = -1e30f;
            float mnew  = fmaxf(mrow[r], half_max(fmaxf(s0, s1)));
            float alpha = __expf(mrow[r] - mnew);
            float p0 = __expf(s0 - mnew), p1 = __expf(s1 - mnew);
            lrow[r] = lrow[r] * alpha + half_sum(p0 + p1);
            mrow[r] = mnew;
#pragma unroll
            for (int ct = 0; ct < 4; ++ct) oacc[ct][r] *= alpha;
            int prow = r + hs * 8;               // stage P (16 rows x 32 keys)
            Plds[prow * 32 + ln]      = (_Float16)p0;
            Plds[prow * 32 + 16 + ln] = (_Float16)p1;
        }
        __syncthreads();

        // --- re-fragment P into A layout: two ds b128 loads per lane ---
        v16h pa = load_a_frag(&Plds[ln * 32], hs);
        __syncthreads();

        // --- O += P @ V over 4 feature col-tiles (contraction = key) ---
#pragma unroll
        for (int ct = 0; ct < 4; ++ct) {
            v16h vb = load_b_frag(vr[ct] + kb, hs);
            oacc[ct] = wmma_f16(pa, vb, oacc[ct]);
        }
    }

#pragma unroll
    for (int ct = 0; ct < 4; ++ct)
#pragma unroll
        for (int r = 0; r < 8; ++r) {
            int row = qt * 16 + r + hs * 8;
            if (row < SEQ)
                aoh[(bb + row) * DIMX + h * DHEAD + ct * 16 + ln] =
                    (_Float16)(oacc[ct][r] / lrow[r]);
        }
}

// ============================================================================
extern "C" void kernel_launch(void* const* d_in, const int* in_sizes, int n_in,
                              void* d_out, int out_size, void* d_ws, size_t ws_size,
                              hipStream_t stream) {
    const float* x = (const float*)d_in[0];
    const unsigned char* mask = (const unsigned char*)d_in[1];
    const float* W[3][4];
    const float* bo[3];
    for (int l = 0; l < 3; ++l) {
        W[l][0] = (const float*)d_in[2 + 5 * l + 0];  // Wq
        W[l][1] = (const float*)d_in[2 + 5 * l + 1];  // Wk
        W[l][2] = (const float*)d_in[2 + 5 * l + 2];  // Wv
        W[l][3] = (const float*)d_in[2 + 5 * l + 3];  // Wo
        bo[l]   = (const float*)d_in[2 + 5 * l + 4];
    }

    float* X = (float*)d_out;                       // running residual stream
    const size_t qe = (size_t)ROWS * DIMX;          // 2,360,320 elements
    char* ws = (char*)d_ws;
    _Float16* qf  = (_Float16*)(ws);                // f16 [row][512]
    _Float16* kf  = (_Float16*)(ws + qe * 2);       // f16 [row][512]
    _Float16* vt  = (_Float16*)(ws + qe * 4);       // f16 [b,h,d][token]
    _Float16* aoh = (_Float16*)(ws + qe * 6);       // f16 [row][512]
    _Float16* xh  = (_Float16*)(ws + qe * 8);       // f16 [row][512]
    _Float16* Wt  = (_Float16*)(ws + qe * 10);      // f16 12x[n][k]  (~6.3 MB)
                                                    // total ws use ~29.9 MB

    hipMemcpyAsync(X, x, qe * sizeof(float), hipMemcpyDeviceToDevice, stream);

    // one-time weight conversion+transpose (12 matrices)
    for (int l = 0; l < 3; ++l)
        for (int i = 0; i < 4; ++i)
            cvt_w_t<<<WELEMS / 256, 256, 0, stream>>>(
                W[l][i], Wt + (size_t)(l * 4 + i) * WELEMS);

    const dim3 gemmGrid(8, (ROWS + 31) / 32);       // qkv: 32 col-tiles/4 waves
    const dim3 projGrid(4, (ROWS + 31) / 32);       // out: 16 col-tiles/4 waves
    const int  cvtBlocks = (int)((qe + 255) / 256);
    for (int l = 0; l < 3; ++l) {
        const _Float16* Wl = Wt + (size_t)l * 4 * WELEMS;
        cvt_x<<<cvtBlocks, 256, 0, stream>>>(X, xh);
        qkv_gemm<<<gemmGrid, 128, 0, stream>>>(xh, Wl, Wl + WELEMS,
                                               Wl + 2 * WELEMS, qf, kf, vt);
        if (l < 2) {
            axial_attn<<<dim3(GHW, HEADS, BATCH), 64, 0, stream>>>(qf, kf, vt,
                                                                   aoh, l);
        } else {
            nearby_attn<<<dim3((SEQ + 15) / 16, HEADS, BATCH), 32, 0, stream>>>(
                qf, kf, vt, mask, aoh);
        }
        out_proj<<<projGrid, 128, 0, stream>>>(aoh, Wl + 3 * WELEMS, bo[l], X);
    }
}
